// STGraphTCN_12987981103579
// MI455X (gfx1250) — matrix operand order repaired and checked by reading, not compiled, not run on hardware
//
#include <hip/hip_runtime.h>
#include <hip/hip_bf16.h>
#include <math.h>

// ---------------- CDNA5 WMMA types / helpers ----------------
typedef __bf16 bf16;
typedef bf16  v16bf __attribute__((ext_vector_type(16)));
typedef float v8f   __attribute__((ext_vector_type(8)));

static constexpr int kB  = 2;
static constexpr int kT  = 64;
static constexpr int kN  = 512;
static constexpr int kD  = 64;
static constexpr int kKS = 9;
static constexpr int kKC = kD * kKS;                  // 576 im2col K
static constexpr float kRSQ = 0.99999500003749975f;   // 1/sqrt(1+1e-5)
static constexpr float kLOGIT_SCALE = 1.0f / 24.0f;   // 0.5/(sqrt(64)*1.5)

__device__ __forceinline__ v8f wmma_bf16(v16bf a, v16bf b, v8f c) {
  // D = A(16x32 bf16) * B(32x16 bf16) + C(16x16 f32)
  return __builtin_amdgcn_wmma_f32_16x16x32_bf16(false, a, false, b, (short)0, c,
                                                 false, false);
}

// A fragment (16x32) from row-major bf16 [M][K]; element(M=r, K) = p[r*ld + K]
// ISA layout: grp0: K 0..7 (v0-3) / 16..23 (v4-7); grp1: 8..15 / 24..31
__device__ __forceinline__ v16bf ldA_rm(const bf16* p, int ld, int lane) {
  int r = lane & 15, grp = lane >> 4;
  v16bf a;
#pragma unroll
  for (int j = 0; j < 8; ++j) {
    int kk = ((j & 4) ? 16 : 0) + grp * 8 + (j & 3) * 2;
    a[2 * j]     = p[r * ld + kk];
    a[2 * j + 1] = p[r * ld + kk + 1];
  }
  return a;
}

// Same A fragment but source is f32 (converted on the fly)
__device__ __forceinline__ v16bf ldA_f32(const float* p, int ld, int lane) {
  int r = lane & 15, grp = lane >> 4;
  v16bf a;
#pragma unroll
  for (int j = 0; j < 8; ++j) {
    int kk = ((j & 4) ? 16 : 0) + grp * 8 + (j & 3) * 2;
    a[2 * j]     = (bf16)p[r * ld + kk];
    a[2 * j + 1] = (bf16)p[r * ld + kk + 1];
  }
  return a;
}

// B fragment for X*Y^T: B(K, N=r) = rows[r*ld + K]  (contiguous 16B runs per
// lane -> the compiler vectorizes into b128 loads)
__device__ __forceinline__ v16bf ldB_rmT(const bf16* rows, int ld, int lane) {
  int r = lane & 15, grp = lane >> 4;
  v16bf b;
#pragma unroll
  for (int j = 0; j < 8; ++j) {
    int kk = ((j & 4) ? 16 : 0) + grp * 8 + (j & 3) * 2;
    b[2 * j]     = rows[r * ld + kk];
    b[2 * j + 1] = rows[r * ld + kk + 1];
  }
  return b;
}

__device__ __forceinline__ float gelu_exact(float x) {
  return 0.5f * x * (1.0f + erff(x * 0.70710678118654752f));
}

// ---------------- workspace layout (bytes) ----------------
static constexpr size_t OFF_HB0   = 0;                       // 1024*64*64 f32
static constexpr size_t OFF_HB1   = 16777216;
static constexpr size_t OFF_HF32  = 33554432;                // (B,T,N,D) f32
static constexpr size_t OFF_HBF   = 50331648;                // (B,T,N,D) bf16
static constexpr size_t OFF_QBF   = 58720256;                // bf16
static constexpr size_t OFF_KBF   = 67108864;                // bf16
static constexpr size_t OFF_CWBF  = 75497472;                // 4*64*576 bf16
static constexpr size_t OFF_QWT   = 75792384;                // 64*64 bf16 (transposed)
static constexpr size_t OFF_KWT   = 75800576;
static constexpr size_t OFF_LWT   = 75808768;
static constexpr size_t OFF_H2L   = 75816960;                // (B*N, D) f32
static constexpr size_t OFF_HBFT  = 76079104;                // (B,T,D,N) bf16

// output layout (f32 elements)
static constexpr size_t OUT_MU    = 0;          // B*N      = 1024
static constexpr size_t OUT_LOGV  = 1024;       // B*N
static constexpr size_t OUT_AFUSE = 2048;       // B*N*N    = 524288
static constexpr size_t OUT_ADYN  = 526336;     // B*T*N*N  = 33554432

// TCN LDS layout (dynamic, 104KB)
static constexpr int LDS_XF    = 0;                // 64*64 f32   (16KB)
static constexpr int LDS_XS    = 16384;            // 64*128 bf16 (16KB)
static constexpr int LDS_PANEL = 32768;            // 64*576 bf16 (72KB)
static constexpr int TCN_LDS_BYTES = 32768 + kT * kKC * 2;   // 106496

// ---------------- kernels ----------------
__global__ void k_prep_weights(const float* conv_w, const float* q_w,
                               const float* k_w, const float* lin_w,
                               bf16* cw, bf16* qwT, bf16* kwT, bf16* lwT) {
  int idx = blockIdx.x * blockDim.x + threadIdx.x;
  const int CW = 4 * kD * kD * kKS;  // 147456
  if (idx < CW) {
    cw[idx] = (bf16)conv_w[idx];
  } else {
    int r = idx - CW;
    int sel = r >> 12, e = r & 4095;
    int c = e >> 6, d = e & 63;          // source [c][d] -> dest [d][c]
    if (sel == 0) qwT[d * kD + c] = (bf16)q_w[e];
    else if (sel == 1) kwT[d * kD + c] = (bf16)k_w[e];
    else if (sel == 2) lwT[d * kD + c] = (bf16)lin_w[e];
  }
}

// h0[bn][d][t] = x[b,t,n,0] * in_w[d] + in_b[d]
__global__ void k_embed(const float* x, const float* in_w, const float* in_b,
                        float* hb0) {
  int idx = blockIdx.x * blockDim.x + threadIdx.x;
  int bn = idx >> 12, rem = idx & 4095;
  int d = rem >> 6, t = rem & 63;
  int b = bn >> 9, n = bn & 511;
  float v = x[(b * kT + t) * kN + n];
  hb0[idx] = v * in_w[d] + in_b[d];
}

// One dilated causal conv layer as a 64x64x576 WMMA GEMM per sequence.
// Residual tile comes in via GLOBAL_LOAD_ASYNC_TO_LDS (ASYNCcnt), the im2col
// panel is materialized transposed so all fragments are contiguous b128 runs.
__global__ void k_tcn_layer(const float* __restrict__ hin,
                            float* __restrict__ hout,
                            const bf16* __restrict__ cw,    // [c_out][c_in*9+k]
                            const float* __restrict__ conv_b,
                            const float* __restrict__ bn_g,
                            const float* __restrict__ bn_b, int dil) {
  extern __shared__ char smem[];
  float* xf    = (float*)(smem + LDS_XF);     // residual copy (f32)
  bf16*  xs    = (bf16*)(smem + LDS_XS);      // [c][64(pad)+64] causal pad
  bf16*  panel = (bf16*)(smem + LDS_PANEL);   // [t][576] im2col bf16

  const int bn = blockIdx.x;
  const int tid = threadIdx.x;
  const int lane = tid & 31, w = tid >> 5;

  // ---- async global->LDS copy of the f32 tile (CDNA5 ASYNCcnt path) ----
  {
    const float* src = hin + bn * 4096 + tid * 4;      // 16B per lane
    unsigned lds_off = (unsigned)LDS_XF + (unsigned)tid * 16u;
#pragma unroll
    for (int it = 0; it < 4; ++it) {                   // 4 * 256 * 16B = 16KB
      asm volatile("global_load_async_to_lds_b128 %0, %1, off"
                   :: "v"(lds_off), "v"(src) : "memory");
      src += 1024;
      lds_off += 4096u;
    }
    asm volatile("s_wait_asynccnt 0x0" ::: "memory");
  }
  __syncthreads();

  // bf16 causal-padded copy built from LDS
  for (int e = tid; e < kD * kT; e += 256) {
    int c = e >> 6, t = e & 63;
    xs[c * 128 + 64 + t] = (bf16)xf[e];
    xs[c * 128 + t] = (bf16)0.0f;      // zero left pad
  }
  __syncthreads();

  // im2col: panel[t*576 + (ci*9+k)] = x[ci][t - (8-k)*dil]
  for (int e = tid; e < kKC * kT; e += 256) {
    int kk = e >> 6, t = e & 63;
    int ci = kk / kKS;
    int kt = kk - ci * kKS;
    panel[t * kKC + kk] = xs[ci * 128 + 64 + t - (8 - kt) * dil];
  }
  __syncthreads();

  // 16 output tiles of 16x16; 8 waves -> 2 tiles each; K = 576 = 18 chunks
  for (int job = w; job < 16; job += 8) {
    int mi = job >> 2, ni = job & 3;
    v8f acc = {};
#pragma unroll 2
    for (int kc = 0; kc < 18; ++kc) {
      v16bf a = ldA_rm(cw + (mi * 16) * kKC + kc * 32, kKC, lane);
      v16bf bfrag = ldB_rmT(panel + (ni * 16) * kKC + kc * 32, kKC, lane);
      acc = wmma_bf16(a, bfrag, acc);
    }
    // epilogue: +bias, bn affine, gelu, residual gelu
    int r = lane & 15, grp = lane >> 4;
#pragma unroll
    for (int i = 0; i < 8; ++i) {
      int c = mi * 16 + i + grp * 8;
      int t = ni * 16 + r;
      float y = acc[i] + conv_b[c];
      y = bn_g[c] * y * kRSQ + bn_b[c];
      y = gelu_exact(y);
      float xn = gelu_exact(xf[c * 64 + t] + y);
      hout[bn * 4096 + c * 64 + t] = xn;
    }
  }
}

// (B*N, D, T) -> (B,T,N,D) f32 + bf16, and (B,T,D,N) bf16 (for P@h B-side)
__global__ void k_transpose(const float* hb, float* hf, bf16* hbf, bf16* hbfT) {
  int idx = blockIdx.x * blockDim.x + threadIdx.x;
  int bn = idx >> 12, rem = idx & 4095;
  int d = rem >> 6, t = rem & 63;
  int b = bn >> 9, n = bn & 511;
  float v = hb[idx];
  size_t bt = (size_t)(b * kT + t);
  size_t dst = (bt * kN + n) * kD + d;
  hf[dst] = v;
  hbf[dst] = (bf16)v;
  hbfT[(bt * kD + d) * kN + n] = (bf16)v;
}

// Q = h@q_w, K = h@k_w (bf16 out) -- 64 rows per block
__global__ void k_qk(const bf16* __restrict__ hbf, const bf16* __restrict__ qwT,
                     const bf16* __restrict__ kwT, bf16* __restrict__ qo,
                     bf16* __restrict__ ko) {
  const int r0 = blockIdx.x * 64;
  const int lane = threadIdx.x & 31, w = threadIdx.x >> 5;
  for (int job = w; job < 32; job += 8) {
    int mat = job >> 4, tj = job & 15;
    int mi = tj >> 2, ni = tj & 3;
    const bf16* wmat = mat ? kwT : qwT;   // transposed: [d_out][c_in]
    v8f acc = {};
#pragma unroll
    for (int kc = 0; kc < 2; ++kc) {
      v16bf a = ldA_rm(hbf + (size_t)(r0 + mi * 16) * kD + kc * 32, kD, lane);
      v16bf b = ldB_rmT(wmat + (ni * 16) * kD + kc * 32, kD, lane);
      acc = wmma_bf16(a, b, acc);
    }
    bf16* dst = mat ? ko : qo;
    int r = lane & 15, grp = lane >> 4;
#pragma unroll
    for (int i = 0; i < 8; ++i) {
      int row = r0 + mi * 16 + i + grp * 8;
      int col = ni * 16 + r;
      dst[(size_t)row * kD + col] = (bf16)acc[i];
    }
  }
}

// Fused graph attention: symmetric logits + softmax (-> Adyn), and at t=T-1
// also Y = P@h, Z = Y@lin_w, h2 = LN(h + Z + lin_b).
__global__ void k_attn(const bf16* __restrict__ qbf, const bf16* __restrict__ kbf,
                       const bf16* __restrict__ hbfT, const float* __restrict__ hf,
                       const float* __restrict__ Aadj, const float* __restrict__ Mm,
                       const bf16* __restrict__ lwT, const float* __restrict__ lin_b,
                       const float* __restrict__ ln_g, const float* __restrict__ ln_b,
                       float* __restrict__ out, float* __restrict__ h2last) {
  extern __shared__ char smem[];
  float* P  = (float*)smem;                 // 64 x 512 logits/probs
  float* yb = (float*)(smem + 131072);      // 64 x 64
  float* zb = yb + 4096;                    // 64 x 64

  const int bt = blockIdx.x >> 3;           // b*T + t
  const int rb = blockIdx.x & 7;            // 64-row block within N
  const int t = bt & 63, b = bt >> 6;
  const int lane = threadIdx.x & 31, w = threadIdx.x >> 5;

  // prefetch bias rows for this row block (global_prefetch_b8 path)
  __builtin_prefetch(Aadj + (size_t)(rb * 64) * kN, 0, 1);
  __builtin_prefetch(Mm + (size_t)(rb * 64) * kN, 0, 1);

  // ---- phase 1: logits tiles (4 M x 32 N tiles), sym = Q.K^T + K.Q^T ----
  for (int job = w; job < 128; job += 8) {
    int mi = job >> 5, nj = job & 31;
    const size_t rowA = (size_t)(bt * kN + rb * 64 + mi * 16) * kD;
    const size_t rowB = (size_t)(bt * kN + nj * 16) * kD;
    v8f acc = {};
#pragma unroll
    for (int kc = 0; kc < 2; ++kc) {
      v16bf aq = ldA_rm(qbf + rowA + kc * 32, kD, lane);
      v16bf bk = ldB_rmT(kbf + rowB + kc * 32, kD, lane);
      acc = wmma_bf16(aq, bk, acc);
      v16bf ak = ldA_rm(kbf + rowA + kc * 32, kD, lane);
      v16bf bq = ldB_rmT(qbf + rowB + kc * 32, kD, lane);
      acc = wmma_bf16(ak, bq, acc);
    }
    int r = lane & 15, grp = lane >> 4;
#pragma unroll
    for (int i = 0; i < 8; ++i) {
      int lr = mi * 16 + i + grp * 8;       // local row 0..63
      int ig = rb * 64 + lr;                // global node i
      int jg = nj * 16 + r;                 // global node j
      float Av = Aadj[ig * kN + jg];
      float Mv = Mm[ig * kN + jg];
      float v = (ig == jg || Mv <= 0.0f) ? -__builtin_inff()
                                         : acc[i] * kLOGIT_SCALE + Av;
      P[lr * kN + jg] = v;
    }
  }
  __syncthreads();

  // ---- row softmax (one wave per 8 rows), write Adyn (+A_fuse at t=63) ----
  for (int rr = 0; rr < 8; ++rr) {
    int lr = w * 8 + rr;
    int ig = rb * 64 + lr;
    float mx = -__builtin_inff();
    for (int c = lane; c < kN; c += 32) mx = fmaxf(mx, P[lr * kN + c]);
#pragma unroll
    for (int s = 16; s > 0; s >>= 1) mx = fmaxf(mx, __shfl_xor(mx, s, 32));
    float sum = 0.0f;
    for (int c = lane; c < kN; c += 32) {
      float e = __expf(P[lr * kN + c] - mx);
      P[lr * kN + c] = e;
      sum += e;
    }
#pragma unroll
    for (int s = 16; s > 0; s >>= 1) sum += __shfl_xor(sum, s, 32);
    float inv = 1.0f / sum;
    for (int c = lane; c < kN; c += 32) {
      float p = P[lr * kN + c] * inv;
      P[lr * kN + c] = p;
      out[OUT_ADYN + ((size_t)bt * kN + ig) * kN + c] = p;
      if (t == kT - 1)
        out[OUT_AFUSE + ((size_t)(b * kN) + ig) * kN + c] = p;
    }
  }
  __syncthreads();

  if (t != kT - 1) return;   // h2 only consumed at the last timestep

  // ---- phase 2a: Y = P @ h  (64 x 64, K = 512), B from h^T (d-major) ----
  for (int job = w; job < 16; job += 8) {
    int mi = job >> 2, nj = job & 3;
    v8f acc = {};
#pragma unroll 2
    for (int kc = 0; kc < 16; ++kc) {
      v16bf a = ldA_f32(P + mi * 16 * kN + kc * 32, kN, lane);
      v16bf bfrag = ldB_rmT(hbfT + ((size_t)bt * kD + nj * 16) * kN + kc * 32,
                            kN, lane);
      acc = wmma_bf16(a, bfrag, acc);
    }
    int r = lane & 15, grp = lane >> 4;
#pragma unroll
    for (int i = 0; i < 8; ++i)
      yb[(mi * 16 + i + grp * 8) * kD + nj * 16 + r] = acc[i];
  }
  __syncthreads();

  // ---- phase 2b: Z = Y @ lin_w (B from lin_w^T) ----
  for (int job = w; job < 16; job += 8) {
    int mi = job >> 2, nj = job & 3;
    v8f acc = {};
#pragma unroll
    for (int kc = 0; kc < 2; ++kc) {
      v16bf a = ldA_f32(yb + mi * 16 * kD + kc * 32, kD, lane);
      v16bf bfrag = ldB_rmT(lwT + (nj * 16) * kD + kc * 32, kD, lane);
      acc = wmma_bf16(a, bfrag, acc);
    }
    int r = lane & 15, grp = lane >> 4;
#pragma unroll
    for (int i = 0; i < 8; ++i)
      zb[(mi * 16 + i + grp * 8) * kD + nj * 16 + r] = acc[i];
  }
  __syncthreads();

  // ---- h2[:, -1] = LN(h + Z + lin_b) ----
  if (threadIdx.x < 64) {
    int lr = threadIdx.x;
    int node = rb * 64 + lr;
    const float* hrow = hf + ((size_t)bt * kN + node) * kD;
    float s = 0.0f;
    for (int d = 0; d < kD; ++d) s += hrow[d] + zb[lr * kD + d] + lin_b[d];
    float m = s * (1.0f / kD);
    float vs = 0.0f;
    for (int d = 0; d < kD; ++d) {
      float v = hrow[d] + zb[lr * kD + d] + lin_b[d] - m;
      vs += v * v;
    }
    float rinv = rsqrtf(vs * (1.0f / kD) + 1e-5f);
    for (int d = 0; d < kD; ++d) {
      float v = hrow[d] + zb[lr * kD + d] + lin_b[d] - m;
      h2last[((size_t)(b * kN) + node) * kD + d] = v * rinv * ln_g[d] + ln_b[d];
    }
  }
}

// final head: LN(norm) -> gelu(h1) -> h2 -> mu/logv
__global__ void k_head(const float* __restrict__ h2last,
                       const float* __restrict__ norm_g, const float* __restrict__ norm_b,
                       const float* __restrict__ h1_w, const float* __restrict__ h1_b,
                       const float* __restrict__ h2_w, const float* __restrict__ h2_b,
                       float* __restrict__ out) {
  __shared__ float sv[64];
  __shared__ float sg[64];
  __shared__ float stat[2];
  const int row = blockIdx.x;          // b*N + n
  const int d = threadIdx.x;
  sv[d] = h2last[(size_t)row * kD + d];
  __syncthreads();
  if (d == 0) {
    float s = 0.0f;
    for (int k = 0; k < kD; ++k) s += sv[k];
    float m = s * (1.0f / kD);
    float vs = 0.0f;
    for (int k = 0; k < kD; ++k) { float u = sv[k] - m; vs += u * u; }
    stat[0] = m;
    stat[1] = rsqrtf(vs * (1.0f / kD) + 1e-5f);
  }
  __syncthreads();
  float hn = (sv[d] - stat[0]) * stat[1] * norm_g[d] + norm_b[d];
  __syncthreads();
  sv[d] = hn;
  __syncthreads();
  float acc = 0.0f;
  for (int k = 0; k < kD; ++k) acc += sv[k] * h1_w[k * kD + d];
  sg[d] = gelu_exact(acc + h1_b[d]);
  __syncthreads();
  if (d < 2) {
    float o = h2_b[d];
    for (int k = 0; k < kD; ++k) o += sg[k] * h2_w[k * 2 + d];
    out[(d == 0 ? OUT_MU : OUT_LOGV) + row] = o;
  }
}

// ---------------- host launch ----------------
extern "C" void kernel_launch(void* const* d_in, const int* in_sizes, int n_in,
                              void* d_out, int out_size, void* d_ws, size_t ws_size,
                              hipStream_t stream) {
  const float* x      = (const float*)d_in[0];
  const float* Aadj   = (const float*)d_in[1];
  const float* Mm     = (const float*)d_in[2];
  const float* in_w   = (const float*)d_in[3];
  const float* in_b   = (const float*)d_in[4];
  const float* conv_w = (const float*)d_in[5];
  const float* conv_b = (const float*)d_in[6];
  const float* bn_g   = (const float*)d_in[7];
  const float* bn_b   = (const float*)d_in[8];
  const float* q_w    = (const float*)d_in[9];
  const float* k_w    = (const float*)d_in[10];
  const float* lin_w  = (const float*)d_in[11];
  const float* lin_b  = (const float*)d_in[12];
  const float* ln_g   = (const float*)d_in[13];
  const float* ln_b   = (const float*)d_in[14];
  const float* norm_g = (const float*)d_in[15];
  const float* norm_b = (const float*)d_in[16];
  const float* h1_w   = (const float*)d_in[17];
  const float* h1_b   = (const float*)d_in[18];
  const float* h2_w   = (const float*)d_in[19];
  const float* h2_b   = (const float*)d_in[20];

  char* ws = (char*)d_ws;
  float* hb0   = (float*)(ws + OFF_HB0);
  float* hb1   = (float*)(ws + OFF_HB1);
  float* hf    = (float*)(ws + OFF_HF32);
  bf16*  hbf   = (bf16*)(ws + OFF_HBF);
  bf16*  qbf   = (bf16*)(ws + OFF_QBF);
  bf16*  kbf   = (bf16*)(ws + OFF_KBF);
  bf16*  cwbf  = (bf16*)(ws + OFF_CWBF);
  bf16*  qwT   = (bf16*)(ws + OFF_QWT);
  bf16*  kwT   = (bf16*)(ws + OFF_KWT);
  bf16*  lwT   = (bf16*)(ws + OFF_LWT);
  float* h2l   = (float*)(ws + OFF_H2L);
  bf16*  hbfT  = (bf16*)(ws + OFF_HBFT);
  float* out   = (float*)d_out;

  // allow large dynamic LDS (CDNA5: 320KB per WGP)
  (void)hipFuncSetAttribute((const void*)k_attn,
                            hipFuncAttributeMaxDynamicSharedMemorySize, 163840);
  (void)hipFuncSetAttribute((const void*)k_tcn_layer,
                            hipFuncAttributeMaxDynamicSharedMemorySize,
                            TCN_LDS_BYTES);

  k_prep_weights<<<624, 256, 0, stream>>>(conv_w, q_w, k_w, lin_w,
                                          cwbf, qwT, kwT, lwT);
  k_embed<<<16384, 256, 0, stream>>>(x, in_w, in_b, hb0);

  const float* tin[4]  = {hb0, hb1, hb0, hb1};
  float*       tout[4] = {hb1, hb0, hb1, hb0};
  for (int l = 0; l < 4; ++l) {
    k_tcn_layer<<<1024, 256, TCN_LDS_BYTES, stream>>>(
        tin[l], tout[l], cwbf + (size_t)l * kD * kD * kKS, conv_b + l * kD,
        bn_g + l * kD, bn_b + l * kD, 1 << l);
  }
  k_transpose<<<16384, 256, 0, stream>>>(hb0, hf, hbf, hbfT);
  k_qk<<<1024, 256, 0, stream>>>(hbf, qwT, kwT, qbf, kbf);
  k_attn<<<1024, 256, 163840, stream>>>(qbf, kbf, hbfT, hf, Aadj, Mm, lwT,
                                        lin_b, ln_g, ln_b, out, h2l);
  k_head<<<1024, 64, 0, stream>>>(h2l, norm_g, norm_b, h1_w, h1_b, h2_w, h2_b,
                                  out);
}
